// GCN_53695681135103
// MI455X (gfx1250) — matrix operand order repaired and checked by reading, not compiled, not run on hardware
//
#include <hip/hip_runtime.h>
#include <hip/hip_bf16.h>
#include <math.h>

// ---------------------------------------------------------------------------
// GCN forward (6 layers) for MI455X / gfx1250.
//   Dense GEMMs  : bf16 WMMA (v_wmma_f32_16x16x32_bf16), f32 accumulate.
//   SpMM         : wave-per-edge gather + relaxed agent-scope f32 atomics
//                  (destination buffer is L2-resident on a 192MB L2).
// ---------------------------------------------------------------------------

typedef __bf16 bf16;
typedef __attribute__((ext_vector_type(8)))  __bf16 v8bf;
typedef __attribute__((ext_vector_type(16))) __bf16 v16bf;
typedef __attribute__((ext_vector_type(8)))  float  v8f;

#define N_NODES 100000
#define N_EDGES 3200000

// ---------------------------------------------------------------------------
// f32 -> bf16 elementwise convert (first-layer activations)
__global__ void k_f32_to_bf16(const float* __restrict__ x, bf16* __restrict__ y,
                              long n) {
  long i = (long)blockIdx.x * blockDim.x + threadIdx.x;
  long stride = (long)gridDim.x * blockDim.x;
  for (; i < n; i += stride) y[i] = (bf16)x[i];
}

// ---------------------------------------------------------------------------
// Transpose W [K, dout] f32 -> Wt [dout_pad, K] bf16, zero-padding rows
// dout..dout_pad-1 so the GEMM never needs B-fragment load guards.
__global__ void k_prep_wt(const float* __restrict__ W, bf16* __restrict__ Wt,
                          int K, int dout, int dout_pad) {
  int idx = blockIdx.x * blockDim.x + threadIdx.x;     // over dout_pad * K
  if (idx >= dout_pad * K) return;
  int n = idx / K, k = idx % K;
  float v = (n < dout) ? W[(long)k * dout + n] : 0.0f;
  Wt[idx] = (bf16)v;
}

// ---------------------------------------------------------------------------
// WMMA GEMM: C[N, dout] = A[N, K](bf16) * Wt[dout_pad, K](bf16, row = out col).
// Block = 128 threads = 4 waves. Each wave: one 16-row M tile x 64 N columns
// (4 accumulators, A fragment reused 4x). K must be a multiple of 32.
//
// A fragment layout used here (16-bit A 16x32, contiguous k, not the ISA table order): lane L holds row M = L&15;
//   elements 0..7  -> K = (L>>4)*16 + e       (contiguous 16B)
//   elements 8..15 -> K = (L>>4)*16 + e       (contiguous 16B)
// B fragment layout (16-bit B 32x16): lane L holds column N = L&15;
//   element e -> K = (L>>4)*16 + e            (contiguous 32B, thanks to Wt)
// C/D layout (f32 16x16): lane L, vgpr r -> M = (L>>4)*8 + r, N = L&15.
__global__ __launch_bounds__(128)
void k_gemm_wmma_bf16(const bf16* __restrict__ A, const bf16* __restrict__ Wt,
                      float* __restrict__ C, int Nrows, int K, int dout) {
  const int lane = threadIdx.x & 31;
  const int wave = threadIdx.x >> 5;
  const int mbase = (blockIdx.x * 4 + wave) * 16;
  const int nbase = blockIdx.y * 64;
  if (mbase >= Nrows) return;                 // wave-uniform exit, EXEC stays full

  const int l15 = lane & 15;
  const int half = lane >> 4;

  int arow = mbase + l15;
  if (arow >= Nrows) arow = Nrows - 1;        // clamp (select, no divergence);
  const bf16* ap = A + (long)arow * K;        // stores are guarded below

  const bf16* wp = Wt + (long)(nbase + l15) * K + half * 16;
  const long wstride = 16L * K;               // 16 output columns per B tile

  v8f acc0 = {}, acc1 = {}, acc2 = {}, acc3 = {};

  for (int k0 = 0; k0 < K; k0 += 32) {
    v8bf alo = *(const v8bf*)(ap + k0 + half * 16);
    v8bf ahi = *(const v8bf*)(ap + k0 + half * 16 + 8);
    v16bf a;
#pragma unroll
    for (int i = 0; i < 8; ++i) { a[i] = alo[i]; a[8 + i] = ahi[i]; }

    v16bf b0 = *(const v16bf*)(wp + k0);
    v16bf b1 = *(const v16bf*)(wp + k0 + wstride);
    v16bf b2 = *(const v16bf*)(wp + k0 + 2 * wstride);
    v16bf b3 = *(const v16bf*)(wp + k0 + 3 * wstride);

    acc0 = __builtin_amdgcn_wmma_f32_16x16x32_bf16(false, a, false, b0,
                                                   (short)0, acc0, false, false);
    acc1 = __builtin_amdgcn_wmma_f32_16x16x32_bf16(false, a, false, b1,
                                                   (short)0, acc1, false, false);
    acc2 = __builtin_amdgcn_wmma_f32_16x16x32_bf16(false, a, false, b2,
                                                   (short)0, acc2, false, false);
    acc3 = __builtin_amdgcn_wmma_f32_16x16x32_bf16(false, a, false, b3,
                                                   (short)0, acc3, false, false);
  }

#pragma unroll
  for (int r = 0; r < 8; ++r) {
    int m = mbase + half * 8 + r;
    if (m < Nrows) {
      float* crow = C + (long)m * dout;
      int n0 = nbase + l15;
      if (n0      < dout) crow[n0]      = acc0[r];
      if (n0 + 16 < dout) crow[n0 + 16] = acc1[r];
      if (n0 + 32 < dout) crow[n0 + 32] = acc2[r];
      if (n0 + 48 < dout) crow[n0 + 48] = acc3[r];
    }
  }
}

// ---------------------------------------------------------------------------
__global__ void k_zero_f32(float* __restrict__ p, long n) {
  long i = (long)blockIdx.x * blockDim.x + threadIdx.x;
  long stride = (long)gridDim.x * blockDim.x;
  for (; i < n; i += stride) p[i] = 0.0f;
}

// ---------------------------------------------------------------------------
// SpMM, d == 256: one wave per edge, 8 contiguous f32 per lane (2x float4),
// relaxed agent-scope fp32 atomics -> global_atomic_add_f32 resolving in L2.
__global__ __launch_bounds__(256)
void k_spmm_d256(const float* __restrict__ sup, const int* __restrict__ src,
                 const int* __restrict__ dst, const float* __restrict__ ew,
                 float* __restrict__ out, int E) {
  int gw = (int)((blockIdx.x * blockDim.x + threadIdx.x) >> 5);
  int lane = threadIdx.x & 31;
  int nw = (int)((gridDim.x * blockDim.x) >> 5);
  for (int e = gw; e < E; e += nw) {
    int s = src[e];
    int d = dst[e];
    float w = ew[e];
    if (e + nw < E) {  // prefetch next edge's source row (global_prefetch_b8)
      __builtin_prefetch(sup + (long)src[e + nw] * 256 + lane * 8, 0, 1);
    }
    const float4* sp = (const float4*)(sup + (long)s * 256 + lane * 8);
    float4 v0 = sp[0];
    float4 v1 = sp[1];
    float* op = out + (long)d * 256 + lane * 8;
    __hip_atomic_fetch_add(op + 0, v0.x * w, __ATOMIC_RELAXED, __HIP_MEMORY_SCOPE_AGENT);
    __hip_atomic_fetch_add(op + 1, v0.y * w, __ATOMIC_RELAXED, __HIP_MEMORY_SCOPE_AGENT);
    __hip_atomic_fetch_add(op + 2, v0.z * w, __ATOMIC_RELAXED, __HIP_MEMORY_SCOPE_AGENT);
    __hip_atomic_fetch_add(op + 3, v0.w * w, __ATOMIC_RELAXED, __HIP_MEMORY_SCOPE_AGENT);
    __hip_atomic_fetch_add(op + 4, v1.x * w, __ATOMIC_RELAXED, __HIP_MEMORY_SCOPE_AGENT);
    __hip_atomic_fetch_add(op + 5, v1.y * w, __ATOMIC_RELAXED, __HIP_MEMORY_SCOPE_AGENT);
    __hip_atomic_fetch_add(op + 6, v1.z * w, __ATOMIC_RELAXED, __HIP_MEMORY_SCOPE_AGENT);
    __hip_atomic_fetch_add(op + 7, v1.w * w, __ATOMIC_RELAXED, __HIP_MEMORY_SCOPE_AGENT);
  }
}

// SpMM, generic d (last layer, d == 40): thread per (edge, feature).
__global__ void k_spmm_gen(const float* __restrict__ sup, const int* __restrict__ src,
                           const int* __restrict__ dst, const float* __restrict__ ew,
                           float* __restrict__ out, int E, int d) {
  long idx = (long)blockIdx.x * blockDim.x + threadIdx.x;
  long stride = (long)gridDim.x * blockDim.x;
  long total = (long)E * d;
  for (; idx < total; idx += stride) {
    int e = (int)(idx / d);
    int j = (int)(idx % d);
    float v = sup[(long)src[e] * d + j] * ew[e];
    __hip_atomic_fetch_add(&out[(long)dst[e] * d + j], v,
                           __ATOMIC_RELAXED, __HIP_MEMORY_SCOPE_AGENT);
  }
}

// ---------------------------------------------------------------------------
// h_next(bf16) = relu(agg + b)  — fuses bias, ReLU, and the bf16 conversion
// for the next layer's WMMA input.
__global__ void k_bias_relu_bf16(const float* __restrict__ agg,
                                 const float* __restrict__ b,
                                 bf16* __restrict__ hb, long n, int d) {
  long i = (long)blockIdx.x * blockDim.x + threadIdx.x;
  long stride = (long)gridDim.x * blockDim.x;
  for (; i < n; i += stride) {
    float v = agg[i] + b[i % d];
    hb[i] = (bf16)(v > 0.0f ? v : 0.0f);
  }
}

// ---------------------------------------------------------------------------
// out = log_softmax(agg + b) over 40 classes, one thread per node.
__global__ void k_logsoftmax40(const float* __restrict__ agg,
                               const float* __restrict__ b,
                               float* __restrict__ out, int Nrows) {
  int i = blockIdx.x * blockDim.x + threadIdx.x;
  if (i >= Nrows) return;
  float v[40];
  float m = -3.402823466e+38f;
#pragma unroll
  for (int j = 0; j < 40; ++j) {
    v[j] = agg[(long)i * 40 + j] + b[j];
    m = fmaxf(m, v[j]);
  }
  float s = 0.0f;
#pragma unroll
  for (int j = 0; j < 40; ++j) s += __expf(v[j] - m);
  float ls = __logf(s);
#pragma unroll
  for (int j = 0; j < 40; ++j) out[(long)i * 40 + j] = v[j] - m - ls;
}

// ---------------------------------------------------------------------------
extern "C" void kernel_launch(void* const* d_in, const int* in_sizes, int n_in,
                              void* d_out, int out_size, void* d_ws, size_t ws_size,
                              hipStream_t stream) {
  (void)in_sizes; (void)n_in; (void)out_size; (void)ws_size;

  const float* x = (const float*)d_in[0];
  const float* W[6] = {(const float*)d_in[1], (const float*)d_in[3],
                       (const float*)d_in[5], (const float*)d_in[7],
                       (const float*)d_in[9], (const float*)d_in[11]};
  const float* Bv[6] = {(const float*)d_in[2], (const float*)d_in[4],
                        (const float*)d_in[6], (const float*)d_in[8],
                        (const float*)d_in[10], (const float*)d_in[12]};
  const float* ew   = (const float*)d_in[13];
  const int*   esrc = (const int*)d_in[14];
  const int*   edst = (const int*)d_in[15];
  float* out = (float*)d_out;

  // Workspace layout (all offsets 256B-aligned):
  //   hB : N*512 bf16  activations (bf16 input to WMMA GEMM)  102,400,000 B
  //   C  : N*256 f32   support (GEMM output / SpMM input)     102,400,000 B
  //   D  : N*256 f32   aggregate (SpMM output)                102,400,000 B
  //   Wt : 256*512 bf16 transposed weights                        262,144 B
  char* ws = (char*)d_ws;
  bf16*  hB = (bf16*)(ws);
  float* C  = (float*)(ws + 102400000L);
  float* D  = (float*)(ws + 204800000L);
  bf16*  Wt = (bf16*)(ws + 307200000L);

  const int N = N_NODES, E = N_EDGES;
  const int dins[6]  = {512, 256, 256, 256, 256, 256};
  const int douts[6] = {256, 256, 256, 256, 256, 40};

  // x (f32) -> hB (bf16)
  k_f32_to_bf16<<<2048, 256, 0, stream>>>(x, hB, (long)N * 512);

  const int mtiles = (N + 15) / 16;           // 6250
  const int gx = (mtiles + 3) / 4;            // 4 waves (m-tiles) per block

  for (int L = 0; L < 6; ++L) {
    const int K = dins[L];
    const int dout = douts[L];
    const int pad = (dout + 63) / 64 * 64;    // B-tile padding (64 cols/wave)
    const long nd = (long)N * dout;

    // W -> Wt (bf16, transposed, zero-padded)
    int twt = pad * K;
    k_prep_wt<<<(twt + 255) / 256, 256, 0, stream>>>(W[L], Wt, K, dout, pad);

    // support = h @ W   (WMMA)
    dim3 grid(gx, pad / 64);
    k_gemm_wmma_bf16<<<grid, 128, 0, stream>>>(hB, Wt, C, N, K, dout);

    // agg = scatter-add over edges
    k_zero_f32<<<2048, 256, 0, stream>>>(D, nd);
    if (dout == 256)
      k_spmm_d256<<<2048, 256, 0, stream>>>(C, esrc, edst, ew, D, E);
    else
      k_spmm_gen<<<4096, 256, 0, stream>>>(C, esrc, edst, ew, D, E, dout);

    if (L < 5)
      k_bias_relu_bf16<<<2048, 256, 0, stream>>>(D, Bv[L], hB, nd, dout);
    else
      k_logsoftmax40<<<(N + 255) / 256, 256, 0, stream>>>(D, Bv[L], out, N);
  }
}
